// LSTMCellModel_76424648065290
// MI455X (gfx1250) — matrix operand order repaired and checked by reading, not compiled, not run on hardware
//
#include <hip/hip_runtime.h>

// ---------------------------------------------------------------------------
// 2-layer LSTM (H=512), B=1024, T=64 obs + 31 generative steps.
// Mixed precision: v_wmma_f32_16x16x32_f16, f32 accumulation, f32 cell state.
// Persistent batch-parallel kernel: 32 workgroups x 32 batch rows, 512 thr.
// Gates are produced/consumed in registers (i,f,g,o tiles per wave); h is
// double-buffered f16 in LDS (compile-time ping/pong so loads stay DS ops);
// cell state in padded LDS.  K-loop unroll capped to avoid VGPR spills.
// ---------------------------------------------------------------------------

typedef __attribute__((ext_vector_type(16))) _Float16 v16h;
typedef __attribute__((ext_vector_type(8)))  _Float16 v8h;
typedef __attribute__((ext_vector_type(8)))  float    v8f;

#define BT   32            // batch rows per workgroup
#define MT   2             // 16-row M tiles per workgroup (BT/16)
#define HH   512           // hidden size (K)
#define HS   520           // padded f16 row stride for h buffers
#define CS   516           // padded f32 row stride for c buffers
#define NTHREADS 512       // 16 wave32s
#define JT   2             // j-tiles (16 cols of H) per wave: 32 tiles / 16 waves

static constexpr size_t SM_HBUF = (size_t)BT * HS * sizeof(_Float16);   // 33280 B
static constexpr size_t SM_CBUF = (size_t)BT * CS * sizeof(float);      // 66048 B
static constexpr size_t SMEM_BYTES = 4 * SM_HBUF + 2 * SM_CBUF
                                   + (size_t)BT * 2 * sizeof(float);    // 265472 B

// ---------------------------------------------------------------------------
// Prologue: convert the three [2048 x 512] f32 weight matrices to f16 in ws.
// ---------------------------------------------------------------------------
__global__ void convert_weights_f16(const float* __restrict__ a,
                                    const float* __restrict__ b,
                                    const float* __restrict__ c,
                                    _Float16* __restrict__ oa,
                                    _Float16* __restrict__ ob,
                                    _Float16* __restrict__ oc,
                                    int n) {
    int i = blockIdx.x * blockDim.x + threadIdx.x;
    if (i < n) {
        oa[i] = (_Float16)a[i];
        ob[i] = (_Float16)b[i];
        oc[i] = (_Float16)c[i];
    }
}

__device__ __forceinline__ v8f wmma16(v16h a, v16h b, v8f c) {
    return __builtin_amdgcn_wmma_f32_16x16x32_f16(false, a, false, b,
                                                  (short)0, c, false, false);
}

// A fragment: halves [k0+aoff, +8) and [k0+16+aoff, +8)
__device__ __forceinline__ v16h ldfragA(const _Float16* p) {
    v8h lo = *(const v8h*)(p);
    v8h hi = *(const v8h*)(p + 16);
    return __builtin_shufflevector(lo, hi,
           0, 1, 2, 3, 4, 5, 6, 7, 8, 9, 10, 11, 12, 13, 14, 15);
}
// B fragment: halves [k0+bko, +8) and [k0+16+bko, +8)
__device__ __forceinline__ v16h ldfragB(const _Float16* p) {
    v8h lo = *(const v8h*)(p);
    v8h hi = *(const v8h*)(p + 16);
    return __builtin_shufflevector(lo, hi,
           0, 1, 2, 3, 4, 5, 6, 7, 8, 9, 10, 11, 12, 13, 14, 15);
}

__device__ __forceinline__ float sigf(float v) {
    return 1.0f / (1.0f + __expf(-v));
}

// One K=512 pass: acc[mt][g] += hTile(32x512,LDS) @ W[g*512+jb]^T
// A fragments (one per M tile) shared across 4 gates; B shared across M.
// Unroll capped at 4 to bound live B fragments (no spills).
__device__ __forceinline__ void gemm_pass(v8f (&acc)[MT][4],
                                          const _Float16* __restrict__ hbuf,
                                          const _Float16* __restrict__ W,
                                          int jb, int lane) {
    const int am   = lane & 15;
    const int aoff = (lane >> 4) * 8;
    const int bko  = (lane >> 4) * 8;
    const _Float16* a0 = hbuf + am * HS + aoff;
    const _Float16* a1 = hbuf + (16 + am) * HS + aoff;
    const _Float16* w0 = W + (size_t)jb * HH + bko;
#pragma unroll 4
    for (int k0 = 0; k0 < HH; k0 += 32) {
        v16h A0 = ldfragA(a0 + k0);
        v16h A1 = ldfragA(a1 + k0);
#pragma unroll
        for (int g = 0; g < 4; ++g) {
            v16h B = ldfragB(w0 + (size_t)g * 512 * HH + k0);
            acc[0][g] = wmma16(A0, B, acc[0][g]);
            acc[1][g] = wmma16(A1, B, acc[1][g]);
        }
    }
}

// ---------------------------------------------------------------------------
// Persistent LSTM kernel.  Workgroup owns batch rows [wg*32, wg*32+32).
// ---------------------------------------------------------------------------
__global__ __launch_bounds__(NTHREADS, 1)
void lstm_persistent(const float* __restrict__ x,        // [B, 2T]
                     const float* __restrict__ Wih1,     // [2048, 2] f32
                     const float* __restrict__ b1,       // [2048]
                     const float* __restrict__ b2,       // [2048]
                     const float* __restrict__ Wl,       // [2, 512]
                     const float* __restrict__ bl,       // [2]
                     const int*   __restrict__ futp,     // scalar (device)
                     const _Float16* __restrict__ Whh1h, // [2048, 512] f16
                     const _Float16* __restrict__ Wih2h, // [2048, 512] f16
                     const _Float16* __restrict__ Whh2h, // [2048, 512] f16
                     float* __restrict__ out,            // [B, 2*(T+future-1)]
                     int T) {
    extern __shared__ char smem[];
    // Compile-time-constant carves (keeps address-space inference -> ds ops).
    _Float16* hA1 = (_Float16*)(smem);
    _Float16* hB1 = (_Float16*)(smem + SM_HBUF);
    _Float16* hA2 = (_Float16*)(smem + 2 * SM_HBUF);
    _Float16* hB2 = (_Float16*)(smem + 3 * SM_HBUF);
    float*    c1  = (float*)(smem + 4 * SM_HBUF);
    float*    c2  = (float*)(smem + 4 * SM_HBUF + SM_CBUF);
    float*    xcur = (float*)(smem + 4 * SM_HBUF + 2 * SM_CBUF);  // [32][2]

    const int tid  = threadIdx.x;
    const int lane = tid & 31;
    const int wave = tid >> 5;
    const int wg   = blockIdx.x;

    const int future = *futp;
    const int nsteps = T + future - 1;
    const int ncols  = 2 * nsteps;

    // Zero all LDS each launch (h, c, xcur) -> deterministic.
    for (size_t i = tid; i < SMEM_BYTES / 4; i += NTHREADS)
        ((uint32_t*)smem)[i] = 0u;
    __syncthreads();

    const int nlo = lane & 15;
    const int mb  = (lane >> 4) * 8;

    // One full LSTM step; h buffers passed as compile-time-constant pointers.
    auto step_body = [&](int step,
                         const _Float16* h1c, _Float16* h1n,
                         const _Float16* h2c, _Float16* h2n) {
        // ----- input: observed x for step < T, else xcur = previous out
        if (step < T) {
            if (tid < BT * 2) {
                int m = tid >> 1, d = tid & 1;
                xcur[m * 2 + d] = x[(size_t)(wg * BT + m) * (2 * T) + 2 * step + d];
            }
            __syncthreads();
        }

        // ===== layer 1: gates in registers, fused cell update
        for (int jt = 0; jt < JT; ++jt) {
            const int jb = ((wave * JT + jt) << 4) + nlo;   // 0..511
            v8f acc[MT][4] = {};
            gemm_pass(acc, h1c, Whh1h, jb, lane);

            const float bi = b1[jb],        bf = b1[jb + 512];
            const float bg = b1[jb + 1024], bo = b1[jb + 1536];
            const float wi0 = Wih1[2 * jb],          wi1 = Wih1[2 * jb + 1];
            const float wf0 = Wih1[2 * (jb + 512)],  wf1 = Wih1[2 * (jb + 512) + 1];
            const float wg0 = Wih1[2 * (jb + 1024)], wg1 = Wih1[2 * (jb + 1024) + 1];
            const float wo0 = Wih1[2 * (jb + 1536)], wo1 = Wih1[2 * (jb + 1536) + 1];
#pragma unroll
            for (int mt = 0; mt < MT; ++mt) {
#pragma unroll
                for (int r = 0; r < 8; ++r) {
                    const int m = mt * 16 + mb + r;
                    const float x0 = xcur[m * 2], x1 = xcur[m * 2 + 1];
                    float gi = acc[mt][0][r] + bi + x0 * wi0 + x1 * wi1;
                    float gf = acc[mt][1][r] + bf + x0 * wf0 + x1 * wf1;
                    float gg = acc[mt][2][r] + bg + x0 * wg0 + x1 * wg1;
                    float go = acc[mt][3][r] + bo + x0 * wo0 + x1 * wo1;
                    float* cp = c1 + m * CS + jb;            // exclusive per lane
                    float cn = sigf(gf) * (*cp) + sigf(gi) * tanhf(gg);
                    *cp = cn;
                    h1n[m * HS + jb] = (_Float16)(sigf(go) * tanhf(cn));
                }
            }
        }
        __syncthreads();   // h1n complete before layer 2 reads it

        // ===== layer 2: h1n@Wih2^T + h2c@Whh2^T, fused update
        for (int jt = 0; jt < JT; ++jt) {
            const int jb = ((wave * JT + jt) << 4) + nlo;
            v8f acc[MT][4] = {};
            gemm_pass(acc, h1n, Wih2h, jb, lane);
            gemm_pass(acc, h2c, Whh2h, jb, lane);

            const float bi = b2[jb],        bf = b2[jb + 512];
            const float bg = b2[jb + 1024], bo = b2[jb + 1536];
#pragma unroll
            for (int mt = 0; mt < MT; ++mt) {
#pragma unroll
                for (int r = 0; r < 8; ++r) {
                    const int m = mt * 16 + mb + r;
                    float gi = acc[mt][0][r] + bi;
                    float gf = acc[mt][1][r] + bf;
                    float gg = acc[mt][2][r] + bg;
                    float go = acc[mt][3][r] + bo;
                    float* cp = c2 + m * CS + jb;
                    float cn = sigf(gf) * (*cp) + sigf(gi) * tanhf(gg);
                    *cp = cn;
                    h2n[m * HS + jb] = (_Float16)(sigf(go) * tanhf(cn));
                }
            }
        }
        __syncthreads();   // h2n complete before output projection

        // ===== output: out[m,d] = h2n[m,:] . Wl[d,:] + bl[d]
        {
            int m   = tid >> 4;         // 0..31
            int d   = (tid >> 3) & 1;   // 0..1
            int seg = tid & 7;          // 0..7  -> 64 elements each
            const _Float16* hr = h2n + m * HS + seg * 64;
            const float*    wl = Wl + d * HH + seg * 64;
            float s = 0.0f;
#pragma unroll 8
            for (int k = 0; k < 64; ++k) s += (float)hr[k] * wl[k];
            s += __shfl_down(s, 4, 32);
            s += __shfl_down(s, 2, 32);
            s += __shfl_down(s, 1, 32);
            if (seg == 0) {
                float v = s + bl[d];
                out[(size_t)(wg * BT + m) * ncols + 2 * step + d] = v;
                xcur[m * 2 + d] = v;   // next step's input in generative phase
            }
        }
        __syncthreads();
    };

    // Step loop unrolled by 2 so ping/pong buffers are compile-time constants.
    int step = 0;
    for (; step + 1 < nsteps; step += 2) {
        step_body(step,     hA1, hB1, hA2, hB2);
        step_body(step + 1, hB1, hA1, hB2, hA2);
    }
    if (step < nsteps)
        step_body(step, hA1, hB1, hA2, hB2);
}

// ---------------------------------------------------------------------------
// Host entry.  Inputs (setup_inputs order):
//  0:x [B,2T] f32   1:Wih1 [2048,2]  2:Whh1 [2048,512]  3:b1 [2048]
//  4:Wih2 [2048,512] 5:Whh2 [2048,512] 6:b2 [2048]
//  7:Wl [2,512]      8:bl [2]          9:future (int scalar, device)
// ---------------------------------------------------------------------------
extern "C" void kernel_launch(void* const* d_in, const int* in_sizes, int n_in,
                              void* d_out, int out_size, void* d_ws, size_t ws_size,
                              hipStream_t stream) {
    const float* x    = (const float*)d_in[0];
    const float* Wih1 = (const float*)d_in[1];
    const float* Whh1 = (const float*)d_in[2];
    const float* b1   = (const float*)d_in[3];
    const float* Wih2 = (const float*)d_in[4];
    const float* Whh2 = (const float*)d_in[5];
    const float* b2   = (const float*)d_in[6];
    const float* Wl   = (const float*)d_in[7];
    const float* bl   = (const float*)d_in[8];
    const int*   fut  = (const int*)d_in[9];
    float*       out  = (float*)d_out;

    const int T = 64;                       // x is [B, 2T]
    const int B = in_sizes[0] / (2 * T);    // 1024

    // f16 weight copies in workspace (3 x 2 MB).
    _Float16* Whh1h = (_Float16*)d_ws;
    _Float16* Wih2h = Whh1h + (size_t)2048 * 512;
    _Float16* Whh2h = Wih2h + (size_t)2048 * 512;

    const int nw = 2048 * 512;
    convert_weights_f16<<<(nw + 255) / 256, 256, 0, stream>>>(
        Whh1, Wih2, Whh2, Whh1h, Wih2h, Whh2h, nw);

    lstm_persistent<<<dim3(B / BT), dim3(NTHREADS), SMEM_BYTES, stream>>>(
        x, Wih1, b1, b2, Wl, bl, fut, Whh1h, Wih2h, Whh2h, out, T);
}
